// Attention_46926812676620
// MI455X (gfx1250) — matrix operand order repaired and checked by reading, not compiled, not run on hardware
//
#include <hip/hip_runtime.h>
#include <cstdint>

// ---------------------------------------------------------------------------
// MI455X (gfx1250) Transformer-XL relative attention, bf16 WMMA pipeline.
//
// Roofline: ~38 GFLOP total, ~100 MB moved -> at 23.3 TB/s the layer is
// latency/bandwidth bound, so everything is cast once to bf16 (halving
// traffic) and all contractions run on v_wmma_f32_16x16x32_bf16 with fp32
// accumulation.  All operands are arranged K-contiguous ("NT" GEMM) so every
// WMMA fragment load is two 16-byte loads per lane.
// Attention band structure: query n attends only to t in [n, n+256]; the
// relative-position gather only touches Rw rows {768..1023, 0}, so term_b is
// a 17-tile GEMM (B~ trick) instead of a dense 1024-wide product.
// ---------------------------------------------------------------------------

typedef unsigned short u16;
typedef __attribute__((ext_vector_type(16))) __bf16 v16bf;
typedef __attribute__((ext_vector_type(8)))  float  v8f;

struct Q2 { uint4 a, b; };

__device__ __forceinline__ v16bf make_frag(uint4 x, uint4 y) {
  Q2 q{ x, y };
  return __builtin_bit_cast(v16bf, q);
}

__device__ __forceinline__ u16 f2bf(float f) {
  uint32_t u = __builtin_bit_cast(uint32_t, f);
  uint32_t r = (u + 0x7FFFu + ((u >> 16) & 1u)) >> 16;   // round-nearest-even
  return (u16)r;
}
__device__ __forceinline__ float bf2f(u16 s) {
  return __builtin_bit_cast(float, (uint32_t)s << 16);
}

// A fragment: 16x32 bf16 (MxK), source row-major with K contiguous, stride ld.
// ISA layout: lanes 0-15 (M=lane): e0..7=K0..7, e8..15=K16..23;
//             lanes 16-31 (M=lane-16): e0..7=K8..15, e8..15=K24..31.
__device__ __forceinline__ v16bf load_a_frag(const u16* base, int ld, int k0) {
  int lane = threadIdx.x & 31;
  int m = lane & 15, hi = lane >> 4;
  const u16* p = base + (size_t)m * ld + k0 + hi * 8;
  return make_frag(*(const uint4*)p, *(const uint4*)(p + 16));
}

// B fragment: 32x16 bf16 (KxN) built from an N x K row-major source (stride ld).
// ISA layout: lanes 0-15 (N=lane): e0..7=K0..7, e8..15=K16..23; lanes 16-31 (N=lane-16): e0..7=K8..15, e8..15=K24..31.
__device__ __forceinline__ v16bf load_b_frag(const u16* base, int ld, int k0) {
  int lane = threadIdx.x & 31;
  int n = lane & 15, hi = lane >> 4;
  const u16* p = base + (size_t)n * ld + k0 + hi * 8;
  return make_frag(*(const uint4*)p, *(const uint4*)(p + 16));
}

__device__ __forceinline__ v8f wmma_bf16(v16bf a, v16bf b, v8f c) {
  return __builtin_amdgcn_wmma_f32_16x16x32_bf16(false, a, false, b,
                                                 (short)0, c, false, false);
}

// One wave computes a 16x64 fp32 tile: A(16xK) x B(64 rows of NxK source)^T.
__device__ __forceinline__ void gemm16x64(const u16* A, int lda,
                                          const u16* B, int ldb,
                                          int ksteps, v8f acc[4]) {
  for (int k = 0; k < ksteps; ++k) {
    if (k + 1 < ksteps) __builtin_prefetch(A + (size_t)(k + 1) * 32, 0, 0);
    v16bf a = load_a_frag(A, lda, k * 32);
#pragma unroll
    for (int nt = 0; nt < 4; ++nt) {
      v16bf b = load_b_frag(B + (size_t)nt * 16 * ldb, ldb, k * 32);
      acc[nt] = wmma_bf16(a, b, acc[nt]);
    }
  }
}

// ------------------------------ pack kernels -------------------------------

// cat = [h ; x] per batch -> bf16, row = b*512 + t, 1024 cols (K contiguous)
__global__ void k_cat(const float* __restrict__ h, const float* __restrict__ x,
                      u16* __restrict__ out) {
  size_t i = (size_t)blockIdx.x * blockDim.x + threadIdx.x;
  if (i >= (size_t)4096 * 1024) return;
  int c = (int)(i & 1023);
  int row = (int)(i >> 10);
  int t = row & 511, b = row >> 9;
  float v = (t < 256) ? h[((size_t)(b * 256 + t)) * 1024 + c]
                      : x[((size_t)(b * 256 + (t - 256))) * 1024 + c];
  out[i] = f2bf(v);
}

// fp32 (R x C) -> bf16 transposed (C x R): makes weight matrices K-contiguous
__global__ void k_transpose(const float* __restrict__ in, u16* __restrict__ out,
                            int R, int C) {
  size_t i = (size_t)blockIdx.x * blockDim.x + threadIdx.x;
  if (i >= (size_t)R * C) return;
  int r = (int)(i / C), c = (int)(i % C);
  out[(size_t)c * R + r] = f2bf(in[i]);
}

__global__ void k_convert(const float* __restrict__ in, u16* __restrict__ out,
                          int n) {
  size_t i = (size_t)blockIdx.x * blockDim.x + threadIdx.x;
  if (i < (size_t)n) out[i] = f2bf(in[i]);
}

// ------------------------------- QKV GEMM ----------------------------------
// (4096 x 1024) x (1024 x 3072); epilogue scatters into q/k (B,H,T,64) bf16
// and v transposed (B,H,64,544) bf16 so the PV GEMM is NT as well.
__global__ __launch_bounds__(256) void k_qkv(const u16* __restrict__ A,
                                             const u16* __restrict__ Bt,
                                             u16* __restrict__ qb,
                                             u16* __restrict__ kb,
                                             u16* __restrict__ vT) {
  int wave = threadIdx.x >> 5, lane = threadIdx.x & 31;
  int row0 = blockIdx.x * 16;
  int c0 = (blockIdx.y * 8 + wave) * 64;
  v8f acc[4] = {};
  gemm16x64(A + (size_t)row0 * 1024, 1024, Bt + (size_t)c0 * 1024, 1024, 32, acc);

  int nloc = lane & 15, mb = (lane >> 4) * 8;
#pragma unroll
  for (int nt = 0; nt < 4; ++nt) {
#pragma unroll
    for (int r = 0; r < 8; ++r) {
      int col = c0 + nt * 16 + nloc;
      int row = row0 + mb + r;
      int b = row >> 9, t = row & 511;
      int which = col >> 10, rc = col & 1023, hh = rc >> 6, dd = rc & 63;
      u16 val = f2bf(acc[nt][r]);
      size_t bh = (size_t)(b * 16 + hh);
      if (which == 0)      qb[(bh * 512 + t) * 64 + dd] = val;
      else if (which == 1) kb[(bh * 512 + t) * 64 + dd] = val;
      else                 vT[(bh * 64 + dd) * 544 + t] = val;
    }
  }
}

// ------------------------------- Rw GEMM -----------------------------------
// Rw = R @ Wkr : (1024 x 1024) x (1024 x 1024), bf16 out (pos-major, d contig)
__global__ __launch_bounds__(256) void k_rw(const u16* __restrict__ A,
                                            const u16* __restrict__ Bt,
                                            u16* __restrict__ Rw) {
  int wave = threadIdx.x >> 5, lane = threadIdx.x & 31;
  int row0 = blockIdx.x * 16;
  int c0 = (blockIdx.y * 8 + wave) * 64;
  v8f acc[4] = {};
  gemm16x64(A + (size_t)row0 * 1024, 1024, Bt + (size_t)c0 * 1024, 1024, 32, acc);
  int nloc = lane & 15, mb = (lane >> 4) * 8;
#pragma unroll
  for (int nt = 0; nt < 4; ++nt)
#pragma unroll
    for (int r = 0; r < 8; ++r)
      Rw[(size_t)(row0 + mb + r) * 1024 + c0 + nt * 16 + nloc] = f2bf(acc[nt][r]);
}

// term_c[b,h,t] = sum_d u[d] * k[b,h,t,d]
__global__ void k_termc(const u16* __restrict__ kb, const float* __restrict__ u,
                        float* __restrict__ c32) {
  int i = blockIdx.x * blockDim.x + threadIdx.x;   // (b*16+h)*512 + t
  if (i >= 8 * 16 * 512) return;
  const u16* kp = kb + (size_t)i * 64;
  float s = 0.f;
#pragma unroll
  for (int d = 0; d < 64; ++d) s += u[d] * bf2f(kp[d]);
  c32[i] = s;
}

// d~[h][jj] = sum_d v[d]*Rw[j, h*64+d], jj 0..255 -> j=768+jj, jj==256 -> j=0
__global__ void k_dtilde(const u16* __restrict__ Rw, const float* __restrict__ v,
                         float* __restrict__ dts) {
  int h = blockIdx.x, jj = threadIdx.x;
  if (jj >= 257) return;
  int j = (jj < 256) ? (768 + jj) : 0;
  const u16* p = Rw + (size_t)j * 1024 + h * 64;
  float s = 0.f;
#pragma unroll
  for (int d = 0; d < 64; ++d) s += v[d] * bf2f(p[d]);
  dts[h * 288 + jj] = s;
}

// ----------------------------- attention core ------------------------------
// One wave handles 16 query rows of one (b,h).  Band: t in [n0, n0+272).
__global__ __launch_bounds__(32) void k_attn(const u16* __restrict__ qb,
                                             const u16* __restrict__ kb,
                                             const u16* __restrict__ vT,
                                             const u16* __restrict__ Rw,
                                             const float* __restrict__ c32,
                                             const float* __restrict__ dts,
                                             u16* __restrict__ obf) {
  __shared__ __attribute__((aligned(16))) float Bs[16 * 288];  // B~ strip
  __shared__ __attribute__((aligned(16))) float Ss[16 * 288];  // scores
  __shared__ __attribute__((aligned(16))) u16   Ps[16 * 288];  // probs (bf16)

  int lane = threadIdx.x;
  int blk = blockIdx.x;
  int nb = blk & 15, h = (blk >> 4) & 15, b = blk >> 8;
  int n0 = nb * 16;
  int nloc = lane & 15, mb = (lane >> 4) * 8;

  // q A-fragments for rows t = 256+n0 .. +15 (K = 64 -> two fragments)
  const u16* qptr = qb + (((size_t)(b * 16 + h)) * 512 + 256 + n0) * 64;
  v16bf a0 = load_a_frag(qptr, 64, 0);
  v16bf a1 = load_a_frag(qptr, 64, 32);

  // --- B~ = q @ Rw[768:1024 | 0:16]^T  (17 tiles of 16) ---
  for (int jt = 0; jt < 17; ++jt) {
    int j0 = (jt < 16) ? (768 + jt * 16) : 0;
    const u16* bp = Rw + (size_t)j0 * 1024 + h * 64;
    v8f c = {};
    c = wmma_bf16(a0, load_b_frag(bp, 1024, 0), c);
    c = wmma_bf16(a1, load_b_frag(bp, 1024, 32), c);
#pragma unroll
    for (int r = 0; r < 8; ++r)
      Bs[(mb + r) * 288 + jt * 16 + nloc] = c[r];
  }
  __syncthreads();

  // --- score strip S[r, ts], ts = t - n0 in [0, 272) ---
  const u16* kbase = kb + (((size_t)(b * 16 + h)) * 512) * 64;
  const float* cc = c32 + ((size_t)(b * 16 + h)) * 512;
  const float* dth = dts + h * 288;
  for (int tt = 0; tt < 17; ++tt) {
    int t0 = n0 + tt * 16;
    const u16* bp = kbase + (size_t)t0 * 64;
    v8f c = {};
    c = wmma_bf16(a0, load_b_frag(bp, 64, 0), c);
    c = wmma_bf16(a1, load_b_frag(bp, 64, 32), c);
#pragma unroll
    for (int r = 0; r < 8; ++r) {
      int row = mb + r;
      int ts = tt * 16 + nloc;
      int d = ts - row;                 // t - n; visible iff 0 <= d <= 256
      float val;
      if (d >= 0 && d <= 256)
        val = (c[r] + cc[t0 + nloc] + Bs[row * 288 + d] + dth[d]) * 0.125f;
      else
        val = -1e30f;
      Ss[row * 288 + ts] = val;
    }
  }
  __syncthreads();

  // --- fp32 softmax per row, emit normalized probs as bf16 (zero-padded) ---
  if (lane < 16) {
    float* srow = Ss + lane * 288;
    float mx = -1e30f;
    for (int ts = 0; ts < 272; ++ts) mx = fmaxf(mx, srow[ts]);
    float sum = 0.f;
    for (int ts = 0; ts < 272; ++ts) {
      float p = __expf(srow[ts] - mx);
      sum += p;
      srow[ts] = p;
    }
    float inv = 1.0f / sum;
    u16* prow = Ps + lane * 288;
    for (int ts = 0; ts < 288; ++ts)
      prow[ts] = (ts < 272) ? f2bf(srow[ts] * inv) : (u16)0;
  }
  __syncthreads();

  // --- out = P @ val : A from LDS (bf16), B from vT (d rows, t contiguous) ---
  const u16* vbase = vT + ((size_t)(b * 16 + h)) * 64 * 544 + n0;
#pragma unroll
  for (int dt = 0; dt < 4; ++dt) {
    v8f c = {};
    for (int kk = 0; kk < 9; ++kk) {
      v16bf pa = load_a_frag(Ps, 288, kk * 32);
      v16bf vb = load_b_frag(vbase + (size_t)(dt * 16) * 544, 544, kk * 32);
      c = wmma_bf16(pa, vb, c);
    }
#pragma unroll
    for (int r = 0; r < 8; ++r)
      obf[((size_t)b * 256 + n0 + mb + r) * 1024 + h * 64 + dt * 16 + nloc] =
          f2bf(c[r]);
  }
}

// --------------------------- output projection -----------------------------
__global__ __launch_bounds__(256) void k_final(const u16* __restrict__ A,
                                               const u16* __restrict__ Bt,
                                               float* __restrict__ out) {
  int wave = threadIdx.x >> 5, lane = threadIdx.x & 31;
  int row0 = blockIdx.x * 16;
  int c0 = (blockIdx.y * 8 + wave) * 64;
  v8f acc[4] = {};
  gemm16x64(A + (size_t)row0 * 1024, 1024, Bt + (size_t)c0 * 1024, 1024, 32, acc);
  int nloc = lane & 15, mb = (lane >> 4) * 8;
#pragma unroll
  for (int nt = 0; nt < 4; ++nt)
#pragma unroll
    for (int r = 0; r < 8; ++r)
      out[(size_t)(row0 + mb + r) * 1024 + c0 + nt * 16 + nloc] = acc[nt][r];
}

// ------------------------------- launcher ----------------------------------
extern "C" void kernel_launch(void* const* d_in, const int* in_sizes, int n_in,
                              void* d_out, int out_size, void* d_ws,
                              size_t ws_size, hipStream_t stream) {
  const float* x    = (const float*)d_in[0];
  const float* h    = (const float*)d_in[1];
  const float* Wqkv = (const float*)d_in[2];
  const float* Wkr  = (const float*)d_in[3];
  const float* R    = (const float*)d_in[4];
  const float* u    = (const float*)d_in[5];
  const float* v    = (const float*)d_in[6];
  const float* Wout = (const float*)d_in[7];
  float* out = (float*)d_out;
  (void)in_sizes; (void)n_in; (void)out_size; (void)ws_size;

  char* p = (char*)d_ws;
  auto carve = [&](size_t bytes) -> char* {
    char* r = p;
    p += (bytes + 255) & ~(size_t)255;
    return r;
  };
  u16* catbf = (u16*)carve((size_t)4096 * 1024 * 2);
  u16* WqkvT = (u16*)carve((size_t)3072 * 1024 * 2);
  u16* WkrT  = (u16*)carve((size_t)1024 * 1024 * 2);
  u16* WoutT = (u16*)carve((size_t)1024 * 1024 * 2);
  u16* Rbf   = (u16*)carve((size_t)1024 * 1024 * 2);
  u16* Rwbf  = (u16*)carve((size_t)1024 * 1024 * 2);
  u16* qb    = (u16*)carve((size_t)8 * 16 * 512 * 64 * 2);
  u16* kb    = (u16*)carve((size_t)8 * 16 * 512 * 64 * 2);
  u16* vT    = (u16*)carve((size_t)8 * 16 * 64 * 544 * 2);
  u16* obf   = (u16*)carve((size_t)8 * 256 * 1024 * 2);
  float* c32 = (float*)carve((size_t)8 * 16 * 512 * 4);
  float* dts = (float*)carve((size_t)16 * 288 * 4);

  k_cat<<<16384, 256, 0, stream>>>(h, x, catbf);
  k_transpose<<<12288, 256, 0, stream>>>(Wqkv, WqkvT, 1024, 3072);
  k_transpose<<<4096, 256, 0, stream>>>(Wkr, WkrT, 1024, 1024);
  k_transpose<<<4096, 256, 0, stream>>>(Wout, WoutT, 1024, 1024);
  k_convert<<<4096, 256, 0, stream>>>(R, Rbf, 1024 * 1024);

  k_qkv<<<dim3(256, 6), 256, 0, stream>>>(catbf, WqkvT, qb, kb, vT);
  k_rw<<<dim3(64, 2), 256, 0, stream>>>(Rbf, WkrT, Rwbf);

  k_termc<<<128, 256, 0, stream>>>(kb, u, c32);
  k_dtilde<<<16, 288, 0, stream>>>(Rwbf, v, dts);

  k_attn<<<2048, 32, 0, stream>>>(qb, kb, vT, Rwbf, c32, dts, obf);

  k_final<<<dim3(128, 2), 256, 0, stream>>>(obf, WoutT, out);
}
